// convMultiheadAttetion_9165460210066
// MI455X (gfx1250) — matrix operand order repaired and checked by reading, not compiled, not run on hardware
//
#include <hip/hip_runtime.h>

// ---------------- problem constants ----------------
#define BATCH   4
#define CH      64
#define HH      128
#define WW      128
#define NPIX    (HH*WW)        // 16384
#define OCH     (3*CH)         // 192
#define NHEADS  8
#define DHEAD   72             // (CH*9)/NHEADS
#define DPAD    80             // pad d/e to 5 tiles of 16
#define DKPAD   96             // pad contraction dim to 3 chunks of 32
#define KCONV   (CH*9)         // 576
#define EPSV    1e-6f
#define NSLICE  16             // N-split for the kv reduction

// --- CDNA5 async global->LDS copy (gated: falls back to plain copy) ---
#if defined(__has_builtin)
#if __has_builtin(__builtin_amdgcn_global_load_async_to_lds_b128) && \
    __has_builtin(__builtin_amdgcn_s_wait_asynccnt)
#define ASYNC_LDS 1
#endif
#endif
#ifndef ASYNC_LDS
#define ASYNC_LDS 0
#endif

typedef __attribute__((ext_vector_type(16))) __bf16 v16bf;
typedef __attribute__((ext_vector_type(8)))  __bf16 v8bf;
typedef __attribute__((ext_vector_type(8)))  float  v8f;

__device__ __forceinline__ v8f zero8() {
  v8f z;
  for (int i = 0; i < 8; ++i) z[i] = 0.f;
  return z;
}

// A-matrix fragment (16xK=32 bf16), source row-major [row][k], stride in elems.
__device__ __forceinline__ v16bf frag_a(const __bf16* base, int row0, int stride, int kbase) {
  const int lane = threadIdx.x & 31;
  const __bf16* p = base + (size_t)(row0 + (lane & 15)) * stride + kbase + ((lane >> 4) << 3);
  v8bf lo = *reinterpret_cast<const v8bf*>(p);
  v8bf hi = *reinterpret_cast<const v8bf*>(p + 16);
  return __builtin_shufflevector(lo, hi, 0,1,2,3,4,5,6,7,8,9,10,11,12,13,14,15);
}

// B-matrix fragment (K=32 x 16 bf16), source stored TRANSPOSED: [n][k] row-major.
__device__ __forceinline__ v16bf frag_b(const __bf16* base, int col0, int stride, int kbase) {
  const int lane = threadIdx.x & 31;
  const __bf16* p = base + (size_t)(col0 + (lane & 15)) * stride + kbase + ((lane >> 4) << 3);
  v8bf lo = *reinterpret_cast<const v8bf*>(p);
  v8bf hi = *reinterpret_cast<const v8bf*>(p + 16);
  return __builtin_shufflevector(lo, hi, 0,1,2,3,4,5,6,7,8,9,10,11,12,13,14,15);
}

__device__ __forceinline__ v8f wmma_bf16(v16bf a, v16bf b, v8f c) {
  return __builtin_amdgcn_wmma_f32_16x16x32_bf16(false, a, false, b, (short)0, c, false, false);
}

#if ASYNC_LDS
typedef __attribute__((ext_vector_type(4))) int v4i;
typedef __attribute__((address_space(1))) v4i v4i_g;   // global int4
typedef __attribute__((address_space(3))) v4i v4i_l;   // LDS int4
__device__ __forceinline__ void async_copy16(const void* g, void* l) {
  __builtin_amdgcn_global_load_async_to_lds_b128((v4i_g*)g, (v4i_l*)l, 0, 0);
}
#endif

// =====================================================================
// K1: qkv = W_qkv * x + b ; relu on q,k.  GEMM M=192, K=64, N=16384/b.
// grid: BATCH*64 blocks (256 n-cols each), 256 threads.
// =====================================================================
__global__ __launch_bounds__(256) void k_qkv(const float* __restrict__ x,
                                             const float* __restrict__ w_qkv,
                                             const float* __restrict__ b_qkv,
                                             float* __restrict__ qkv) {
  __shared__ __align__(16) __bf16 sW[OCH * CH];     // [o][c] row-major (A)
  __shared__ __align__(16) __bf16 sXT[256 * CH];    // [n][c] transposed (B)
  const int tid = threadIdx.x;
  const int b = blockIdx.x >> 6;
  const int nbase = (blockIdx.x & 63) * 256;

  for (int i = tid; i < OCH * CH; i += 256) sW[i] = (__bf16)w_qkv[i];
  const float* xb = x + (size_t)b * CH * NPIX;
  for (int i = tid; i < 256 * CH; i += 256) {
    int c = i >> 8, n = i & 255;
    const float* gp = xb + c * NPIX + nbase + n;
    __builtin_prefetch(gp + 256, 0, 1);     // global_prefetch_b8
    sXT[n * CH + c] = (__bf16)(*gp);
  }
  __syncthreads();

  const int wave = tid >> 5, lane = tid & 31;
  float* outb = qkv + (size_t)b * OCH * NPIX;
  const int mhi = (lane >> 4) << 3;
  for (int t = wave; t < 12 * 16; t += 8) {
    int ot = t >> 4, nt = t & 15;
    v8f acc = zero8();
    for (int kc = 0; kc < CH; kc += 32) {
      v16bf a  = frag_a(sW,  ot * 16, CH, kc);
      v16bf bb = frag_b(sXT, nt * 16, CH, kc);
      acc = wmma_bf16(a, bb, acc);
    }
    int n = nbase + nt * 16 + (lane & 15);
    for (int v = 0; v < 8; ++v) {
      int o = ot * 16 + v + mhi;
      float r = acc[v] + b_qkv[o];
      if (o < 2 * CH) r = fmaxf(r, 0.f);    // relu on q and k rows
      outb[o * NPIX + n] = r;
    }
  }
}

// =====================================================================
// K2a: partial kv[d][e] = sum_{n in slice} K[d][n] V[e][n] ; partial ksum.
// grid: BATCH*NHEADS*NSLICE = 512 blocks, each slice covers 1024 n.
// =====================================================================
__global__ __launch_bounds__(256) void k_kv_part(const float* __restrict__ qkv,
                                                 float* __restrict__ kvPart,   // [bh*16+sl][80*80]
                                                 float* __restrict__ ksPart) { // [bh*16+sl][80]
  __shared__ __align__(16) __bf16 sK[DPAD * 32];   // [d][n] (A)
  __shared__ __align__(16) __bf16 sV[DPAD * 32];   // [e][n] (B, transposed view)
  const int tid = threadIdx.x, wave = tid >> 5, lane = tid & 31;
  const int sl = blockIdx.x & (NSLICE - 1);
  const int bh = blockIdx.x >> 4;
  const int b = bh >> 3, h = bh & 7;
  const float* kb = qkv + (size_t)b * OCH * NPIX + (size_t)CH * NPIX;
  const float* vb = qkv + (size_t)b * OCH * NPIX + (size_t)2 * CH * NPIX;

  int owned[4]; int nown = 0;                      // 25 kv tiles over 8 waves
  for (int t = wave; t < 25; t += 8) owned[nown++] = t;
  v8f acc[4];
  for (int i = 0; i < 4; ++i) acc[i] = zero8();
  float ks = 0.f;

  for (int ch = sl * 32; ch < (sl + 1) * 32; ++ch) {
    const int nb0 = ch * 32;
    const int y = nb0 >> 7, xb0 = nb0 & 127;
    for (int idx = tid; idx < DPAD * 32; idx += 256) {
      int r = idx >> 5, i = idx & 31;
      float fk = 0.f, fv = 0.f;
      if (r < DHEAD) {
        int c = h * 8 + r / 9, p = r % 9;
        int yy = y + p / 3 - 1, xx = xb0 + i + (p % 3) - 1;
        if (yy >= 0 && yy < HH && xx >= 0 && xx < WW) {
          int off = c * NPIX + yy * WW + xx;
          fk = kb[off];
          fv = vb[off];
        }
      }
      sK[idx] = (__bf16)fk;
      sV[idx] = (__bf16)fv;
    }
    __syncthreads();
    if (tid < DPAD) {
      float s = 0.f;
      for (int i = 0; i < 32; ++i) s += (float)sK[tid * 32 + i];
      ks += s;
    }
    for (int i = 0; i < nown; ++i) {
      int dt = owned[i] / 5, et = owned[i] % 5;
      v16bf a  = frag_a(sK, dt * 16, 32, 0);
      v16bf bb = frag_b(sV, et * 16, 32, 0);
      acc[i] = wmma_bf16(a, bb, acc[i]);
    }
    __syncthreads();
  }

  float* po = kvPart + (size_t)blockIdx.x * (DPAD * DPAD);
  const int mhi = (lane >> 4) << 3;
  for (int i = 0; i < nown; ++i) {
    int dt = owned[i] / 5, et = owned[i] % 5;
    int e = et * 16 + (lane & 15);
    for (int v = 0; v < 8; ++v) {
      int d = dt * 16 + v + mhi;
      po[e * DPAD + d] = acc[i][v];
    }
  }
  if (tid < DPAD) ksPart[blockIdx.x * DPAD + tid] = ks;
}

// =====================================================================
// K2b: reduce 16 slices -> kvT (bf16, [e][DKPAD] with zero pad) + ksum.
// grid: BATCH*NHEADS = 32 blocks, 256 threads. Tiny, L2-resident.
// =====================================================================
__global__ __launch_bounds__(256) void k_kv_reduce(const float* __restrict__ kvPart,
                                                   const float* __restrict__ ksPart,
                                                   __bf16* __restrict__ kvT,
                                                   float* __restrict__ ksum) {
  const int tid = threadIdx.x;
  const int bh = blockIdx.x;
  __bf16* kvo = kvT + (size_t)bh * DPAD * DKPAD;
  for (int idx = tid; idx < DPAD * DKPAD; idx += 256) {
    int e = idx / DKPAD, d = idx % DKPAD;
    float s = 0.f;
    if (d < DPAD) {
      for (int sl = 0; sl < NSLICE; ++sl)
        s += kvPart[(size_t)(bh * NSLICE + sl) * (DPAD * DPAD) + e * DPAD + d];
    }
    kvo[e * DKPAD + d] = (__bf16)s;
  }
  if (tid < DKPAD) {
    float s = 0.f;
    if (tid < DPAD)
      for (int sl = 0; sl < NSLICE; ++sl)
        s += ksPart[(bh * NSLICE + sl) * DPAD + tid];
    ksum[bh * DKPAD + tid] = s;
  }
}

// =====================================================================
// K3: o = (q_unf · kv) * rcp(q_unf·ksum + eps), with 1-row halo in LDS,
// then fold + divide-by-counts -> y0.
// grid: BATCH*NHEADS*32 blocks (4-row bands), 256 threads.
// =====================================================================
#define RROWS 4
__global__ __launch_bounds__(256) void k_attn(const float* __restrict__ qkv,
                                              const __bf16* __restrict__ kvT,
                                              const float* __restrict__ ksum,
                                              float* __restrict__ y0) {
  __shared__ __align__(16) __bf16 sO[(RROWS + 2) * WW * DHEAD]; // o rows incl. halo
  __shared__ __align__(16) __bf16 sQ[WW * DKPAD];               // [x][d] transposed (A)
  __shared__ __align__(16) __bf16 sKV[DPAD * DKPAD];            // [e][d] (B)
  __shared__ float sDen[WW];                                    // reciprocal denominators
  __shared__ float sKs[DHEAD];

  const int tid = threadIdx.x, wave = tid >> 5, lane = tid & 31;
  int bid = blockIdx.x;
  const int yblk = bid & 31; bid >>= 5;
  const int h = bid & 7;     bid >>= 3;
  const int b = bid;
  const int y0r = yblk * RROWS;

  const __bf16* kvo = kvT + (size_t)(b * NHEADS + h) * DPAD * DKPAD;
#if ASYNC_LDS
  for (int i = tid; i < (DPAD * DKPAD) / 8; i += 256)   // 960 x 16-byte async copies
    async_copy16(kvo + i * 8, sKV + i * 8);
  __builtin_amdgcn_s_wait_asynccnt(0);
#else
  for (int i = tid; i < DPAD * DKPAD; i += 256) sKV[i] = kvo[i];
#endif
  if (tid < DHEAD) sKs[tid] = ksum[(b * NHEADS + h) * DKPAD + tid];

  const float* qb = qkv + (size_t)b * OCH * NPIX;   // q rows start at 0
  const int mhi = (lane >> 4) << 3;

  for (int yy = y0r - 1; yy <= y0r + RROWS; ++yy) {
    if (yy < 0 || yy >= HH) continue;               // uniform over block
    __syncthreads();
    for (int idx = tid; idx < WW * DKPAD; idx += 256) {
      int xx = idx / DKPAD, dk = idx % DKPAD;
      float val = 0.f;
      if (dk < DHEAD) {
        int c = h * 8 + dk / 9, p = dk % 9;
        int ys = yy + p / 3 - 1, xs = xx + (p % 3) - 1;
        if (ys >= 0 && ys < HH && xs >= 0 && xs < WW)
          val = qb[c * NPIX + ys * WW + xs];
      }
      sQ[xx * DKPAD + dk] = (__bf16)val;
    }
    __syncthreads();
    if (tid < WW) {
      float s = EPSV;
      for (int dk = 0; dk < DHEAD; ++dk) s += (float)sQ[tid * DKPAD + dk] * sKs[dk];
      sDen[tid] = 1.f / s;
    }
    __syncthreads();
    const int row = yy - (y0r - 1);
    const int nt = wave;                            // 8 waves x 16-wide x strips
    for (int et = 0; et < 5; ++et) {
      v8f acc = zero8();
      for (int kc = 0; kc < DKPAD; kc += 32) {
        v16bf a  = frag_a(sQ,  nt * 16, DKPAD, kc);
        v16bf bb = frag_b(sKV, et * 16, DKPAD, kc);
        acc = wmma_bf16(a, bb, acc);
      }
      int e = et * 16 + (lane & 15);
      if (e < DHEAD) {
        for (int v = 0; v < 8; ++v) {
          int xx = nt * 16 + v + mhi;
          sO[(row * WW + xx) * DHEAD + e] = (__bf16)(acc[v] * sDen[xx]);
        }
      }
    }
  }
  __syncthreads();

  // fold + counts normalize
  float* yb = y0 + (size_t)b * CH * NPIX;
  for (int idx = tid; idx < 8 * RROWS * WW; idx += 256) {
    int cl = idx / (RROWS * WW);
    int rem = idx % (RROWS * WW);
    int yl = rem / WW, xx = rem % WW;
    int y = y0r + yl;
    int nvi = 0, nvj = 0;
    for (int i2 = 0; i2 < 3; ++i2) { int ys = y + 1 - i2; if (ys >= 0 && ys < HH) nvi++; }
    for (int j2 = 0; j2 < 3; ++j2) { int xs = xx + 1 - j2; if (xs >= 0 && xs < WW) nvj++; }
    float s = 0.f;
    for (int i2 = 0; i2 < 3; ++i2) {
      int ys = y + 1 - i2;
      if (ys < 0 || ys >= HH) continue;
      int rrow = ys - (y0r - 1);
      for (int j2 = 0; j2 < 3; ++j2) {
        int xs = xx + 1 - j2;
        if (xs < 0 || xs >= WW) continue;
        s += (float)sO[(rrow * WW + xs) * DHEAD + (cl * 9 + i2 * 3 + j2)];
      }
    }
    yb[(h * 8 + cl) * NPIX + y * WW + xx] = s / (float)(nvi * nvj);
  }
}

// =====================================================================
// K4: 3x3 conv as implicit GEMM: M=co(64), K=576, N=x(128 per row).
// grid: BATCH*HH blocks (one image row), 256 threads.
// =====================================================================
__global__ __launch_bounds__(256) void k_conv(const float* __restrict__ y0,
                                              const float* __restrict__ w_proj,
                                              const float* __restrict__ b_proj,
                                              float* __restrict__ out) {
  __shared__ __align__(16) __bf16 sP[WW * 32];   // [x][k] transposed patch (B)
  __shared__ __align__(16) __bf16 sWc[CH * 32];  // [co][k] (A)
  const int tid = threadIdx.x, wave = tid >> 5, lane = tid & 31;
  const int b = blockIdx.x >> 7;
  const int y = blockIdx.x & 127;
  const float* yb = y0 + (size_t)b * CH * NPIX;

  v8f acc[4];
  for (int i = 0; i < 4; ++i) acc[i] = zero8();
  const int nt = wave;                           // 8 strips of 16 x-cols

  for (int kb = 0; kb < KCONV; kb += 32) {
    __syncthreads();
    for (int idx = tid; idx < WW * 32; idx += 256) {
      int kl = idx >> 7, xx = idx & 127;
      int k = kb + kl;
      int ci = k / 9, p = k % 9;
      int ys = y + p / 3 - 1, xs = xx + (p % 3) - 1;
      float val = 0.f;
      if (ys >= 0 && ys < HH && xs >= 0 && xs < WW) {
        const float* gp = yb + ci * NPIX + ys * WW + xs;
        __builtin_prefetch(gp + WW, 0, 1);
        val = *gp;
      }
      sP[xx * 32 + kl] = (__bf16)val;
    }
    for (int idx = tid; idx < CH * 32; idx += 256) {
      int co = idx >> 5, kl = idx & 31;
      sWc[idx] = (__bf16)w_proj[co * KCONV + kb + kl];
    }
    __syncthreads();
    for (int cot = 0; cot < 4; ++cot) {
      v16bf a  = frag_a(sWc, cot * 16, 32, 0);
      v16bf bb = frag_b(sP,  nt * 16, 32, 0);
      acc[cot] = wmma_bf16(a, bb, acc[cot]);
    }
  }

  float* ob = out + (size_t)b * CH * NPIX;
  const int mhi = (lane >> 4) << 3;
  const int xx = nt * 16 + (lane & 15);
  for (int cot = 0; cot < 4; ++cot) {
    for (int v = 0; v < 8; ++v) {
      int co = cot * 16 + v + mhi;
      ob[co * NPIX + y * WW + xx] = acc[cot][v] + b_proj[co];
    }
  }
}

// =====================================================================
extern "C" void kernel_launch(void* const* d_in, const int* in_sizes, int n_in,
                              void* d_out, int out_size, void* d_ws, size_t ws_size,
                              hipStream_t stream) {
  (void)in_sizes; (void)n_in; (void)out_size; (void)ws_size;
  const float* x      = (const float*)d_in[0];
  const float* w_qkv  = (const float*)d_in[1];
  const float* b_qkv  = (const float*)d_in[2];
  const float* w_proj = (const float*)d_in[3];
  const float* b_proj = (const float*)d_in[4];
  float* out = (float*)d_out;

  // workspace layout (~81 MB total)
  char* ws = (char*)d_ws;
  float* qkv = (float*)ws;                                        // 50.3 MB
  size_t off = (size_t)BATCH * OCH * NPIX * sizeof(float);
  float* kvPart = (float*)(ws + off);                             // 13.1 MB
  off += (size_t)BATCH * NHEADS * NSLICE * DPAD * DPAD * sizeof(float);
  float* ksPart = (float*)(ws + off);                             // 160 KB
  off += (size_t)BATCH * NHEADS * NSLICE * DPAD * sizeof(float);
  __bf16* kvT = (__bf16*)(ws + off);                              // 480 KB
  off += (size_t)BATCH * NHEADS * DPAD * DKPAD * sizeof(__bf16);
  float* ksum = (float*)(ws + off);                               // 12 KB
  off += (size_t)BATCH * NHEADS * DKPAD * sizeof(float);
  float* y0 = (float*)(ws + off);                                 // 16.8 MB

  k_qkv      <<<BATCH * 64,              256, 0, stream>>>(x, w_qkv, b_qkv, qkv);
  k_kv_part  <<<BATCH * NHEADS * NSLICE, 256, 0, stream>>>(qkv, kvPart, ksPart);
  k_kv_reduce<<<BATCH * NHEADS,          256, 0, stream>>>(kvPart, ksPart, kvT, ksum);
  k_attn     <<<BATCH * NHEADS * 32,     256, 0, stream>>>(qkv, kvT, ksum, y0);
  k_conv     <<<BATCH * HH,              256, 0, stream>>>(y0, w_proj, b_proj, out);
}
